// GATModel_63917703299121
// MI455X (gfx1250) — matrix operand order repaired and checked by reading, not compiled, not run on hardware
//
#include <hip/hip_runtime.h>
#include <stdint.h>
#include <stddef.h>

#define N_NODES 100000
#define N_EDGES 1600000
#define E_TOT   (N_EDGES + N_NODES)
#define IN_DIM  128
#define HID     64
#define HEADS   4
#define C1      (HEADS * HID)   // 256

// tilesM = 100000/16 = 6250 = 5 * 1250  ->  5 M-tiles per wave, no remainder
#define MTILES  5

typedef __attribute__((ext_vector_type(16))) _Float16 v16h;
typedef __attribute__((ext_vector_type(8)))  float    v8f;

#define NEG_BIG (-3.402823466e+38f)

__device__ __forceinline__ void atomic_max_f32(float* addr, float v) {
    // sign-split trick; buffers initialized to very-negative
    if (v >= 0.0f) atomicMax((int*)addr, __float_as_int(v));
    else           atomicMin((unsigned int*)addr, __float_as_uint(v));
}

// ---------------------------------------------------------------- edge build
__global__ void k_build_edges(const long long* __restrict__ ei,
                              int* __restrict__ src, int* __restrict__ dst) {
    int t = blockIdx.x * blockDim.x + threadIdx.x;
    if (t >= E_TOT) return;
    if (t < N_EDGES) {
        src[t] = (int)ei[t];
        dst[t] = (int)ei[(size_t)N_EDGES + t];
    } else {
        int n = t - N_EDGES;            // self-loops
        src[t] = n; dst[t] = n;
    }
}

// ---------------------------------------------------------------- converts / fills
__global__ void k_f32_to_f16(const float* __restrict__ in, _Float16* __restrict__ out, size_t n) {
    size_t stride = (size_t)gridDim.x * blockDim.x;
    for (size_t t = blockIdx.x * (size_t)blockDim.x + threadIdx.x; t < n; t += stride)
        out[t] = (_Float16)in[t];
}

__global__ void k_fill_f32(float* __restrict__ p, float v, size_t n) {
    size_t stride = (size_t)gridDim.x * blockDim.x;
    for (size_t t = blockIdx.x * (size_t)blockDim.x + threadIdx.x; t < n; t += stride)
        p[t] = v;
}

// ---------------------------------------------------------------- B pack (f32 -> f16 WMMA fragment layout)
// Bp linear index: (((tn*(K/32) + kt)*32 + lane)*16 + j)
//   lane<16:  element = B[kt*32 +      j][tn*16 + (lane&15)]
//   lane>=16: element = B[kt*32 + 16 + j][tn*16 + (lane&15)]
__global__ void k_pack_b(const float* __restrict__ B, _Float16* __restrict__ Bp,
                         int K, int Nn) {
    int total = K * Nn;
    int t = blockIdx.x * blockDim.x + threadIdx.x;
    if (t >= total) return;
    int j     = t & 15;
    int rest  = t >> 4;
    int lane  = rest & 31;
    int rest2 = rest >> 5;
    int KT    = K >> 5;
    int kt    = rest2 % KT;
    int tn    = rest2 / KT;
    int k   = kt * 32 + ((lane >> 4) << 4) + j;
    int col = tn * 16 + (lane & 15);
    Bp[t] = (_Float16)B[(size_t)k * Nn + col];
}

// ---------------------------------------------------------------- WMMA GEMM
// C[M,Nn] = A[M,K] @ B[K,Nn] (f16 in, f32 out), optional bias + ReLU.
// Block: 256 threads = 8 waves, fixed N-tile tn = blockIdx.x.
// Packed B strip for tn staged in LDS; each wave computes MTILES M-tiles x 16
// cols; requires (M/16) % MTILES == 0 so the inner loop is guard-free:
// per K-step: 1 LDS B-fragment load + MTILES straight-line WMMAs.
template<int K>
__global__ __launch_bounds__(256)
void k_wmma_gemm(const _Float16* __restrict__ A, const _Float16* __restrict__ Bp,
                 const float* __restrict__ bias, float* __restrict__ C,
                 int M, int Nn, int relu_act, int use_bias) {
    constexpr int KT = K / 32;
    __shared__ _Float16 Bs[K * 16] __attribute__((aligned(32)));

    int tilesM = M >> 4;                                // divisible by MTILES
    int tn     = blockIdx.x;

    // ---- cooperative stage of packed B strip (K*16 halves = K*32 bytes) ----
    {
        const int4* gsrc = (const int4*)(Bp + (size_t)tn * (K * 16));
        int4*       ldst = (int4*)Bs;
        int nchunk = (K * 16 * 2) / 16;                 // 16B chunks
        for (int c = threadIdx.x; c < nchunk; c += blockDim.x)
            ldst[c] = gsrc[c];
    }
    __syncthreads();

    int wave = threadIdx.x >> 5;
    int lane = threadIdx.x & 31;
    int tm0  = (blockIdx.y * 8 + wave) * MTILES;        // first of MTILES M-tiles
    if (tm0 >= tilesM) return;                          // wave-uniform padding waves

    int halfsel = lane >> 4;
    int l16     = lane & 15;
    int bcol    = tn * 16 + l16;

    // per-subtile A base pointers (row = (tm0+s)*16 + l16)
    const _Float16* abase[MTILES];
#pragma unroll
    for (int s = 0; s < MTILES; ++s)
        abase[s] = A + (size_t)((tm0 + s) * 16 + l16) * K + halfsel * 16;

    v8f acc[MTILES] = {};
#pragma unroll
    for (int kt = 0; kt < KT; ++kt) {
        v16h b = *reinterpret_cast<const v16h*>(&Bs[(kt * 32 + lane) * 16]);
#pragma unroll
        for (int s = 0; s < MTILES; ++s) {
            const _Float16* ap = abase[s] + kt * 32;
            __builtin_prefetch(ap + 32, 0, 1);          // -> global_prefetch_b8
            v16h a;
#pragma unroll
            for (int i = 0; i < 16; ++i) { a[i] = ap[i]; }
            acc[s] = __builtin_amdgcn_wmma_f32_16x16x32_f16(
                false, a, false, b, (short)0, acc[s], false, false);
        }
    }

    float badd = use_bias ? bias[bcol] : 0.0f;
#pragma unroll
    for (int s = 0; s < MTILES; ++s) {
        int orow0 = (tm0 + s) * 16 + halfsel * 8;
#pragma unroll
        for (int r = 0; r < 8; ++r) {
            float v = acc[s][r] + badd;
            if (relu_act) v = v > 0.0f ? v : 0.0f;
            C[(size_t)(orow0 + r) * Nn + bcol] = v;
        }
    }
}

// ---------------------------------------------------------------- node attention logits
__global__ void k_attn_logits(const float* __restrict__ h, const float* __restrict__ a_src,
                              const float* __restrict__ a_dst, float* __restrict__ es,
                              float* __restrict__ ed, int n, int heads, int dim) {
    int t = blockIdx.x * blockDim.x + threadIdx.x;
    if (t >= n * heads) return;
    int hh = t % heads;
    const float* hp = h + (size_t)t * dim;   // == node*heads*dim + hh*dim
    const float* as = a_src + (size_t)hh * dim;
    const float* ad = a_dst + (size_t)hh * dim;
    float s = 0.f, d = 0.f;
    for (int i = 0; i < dim; ++i) { float v = hp[i]; s += v * as[i]; d += v * ad[i]; }
    es[t] = s; ed[t] = d;
}

// ---------------------------------------------------------------- edge pass 1: logits + segment max
__global__ void k_edge_logit_max(const int* __restrict__ src, const int* __restrict__ dst,
                                 const float* __restrict__ es, const float* __restrict__ ed,
                                 float* __restrict__ elog, float* __restrict__ segmax, int heads) {
    size_t total  = (size_t)E_TOT * heads;
    size_t stride = (size_t)gridDim.x * blockDim.x;
    for (size_t t = blockIdx.x * (size_t)blockDim.x + threadIdx.x; t < total; t += stride) {
        size_t e = t / heads; int hh = (int)(t % heads);
        float x = es[(size_t)src[e] * heads + hh] + ed[(size_t)dst[e] * heads + hh];
        x = x > 0.0f ? x : 0.2f * x;          // leaky_relu 0.2
        elog[t] = x;
        atomic_max_f32(&segmax[(size_t)dst[e] * heads + hh], x);
    }
}

// ---------------------------------------------------------------- edge pass 2: alpha + segment sum
__global__ void k_edge_alpha_denom(const int* __restrict__ dst, float* __restrict__ elog,
                                   const float* __restrict__ segmax, float* __restrict__ denom,
                                   int heads) {
    size_t total  = (size_t)E_TOT * heads;
    size_t stride = (size_t)gridDim.x * blockDim.x;
    for (size_t t = blockIdx.x * (size_t)blockDim.x + threadIdx.x; t < total; t += stride) {
        size_t e = t / heads; int hh = (int)(t % heads);
        size_t di = (size_t)dst[e] * heads + hh;
        float alpha = __expf(elog[t] - segmax[di]);
        elog[t] = alpha;                      // logits -> alpha in place
        atomicAdd(&denom[di], alpha);
    }
}

// ---------------------------------------------------------------- edge pass 3: weighted scatter-add
__global__ void k_edge_aggregate(const int* __restrict__ src, const int* __restrict__ dst,
                                 const float* __restrict__ alpha, const float* __restrict__ denom,
                                 const float* __restrict__ h, float* __restrict__ agg,
                                 int heads, int dim) {
    int Cc = heads * dim;
    size_t total  = (size_t)E_TOT * Cc;
    size_t stride = (size_t)gridDim.x * blockDim.x;
    for (size_t t = blockIdx.x * (size_t)blockDim.x + threadIdx.x; t < total; t += stride) {
        size_t e = t / Cc; int c = (int)(t % Cc); int hh = c / dim;
        int dnode = dst[e];
        float w = alpha[e * heads + hh] / (denom[(size_t)dnode * heads + hh] + 1e-16f);
        atomicAdd(&agg[(size_t)dnode * Cc + c], h[(size_t)src[e] * Cc + c] * w);
    }
}

// ---------------------------------------------------------------- bias + ELU -> f16
__global__ void k_bias_elu_f16(const float* __restrict__ in, const float* __restrict__ b,
                               _Float16* __restrict__ out, size_t n, int Cc) {
    size_t total  = n * (size_t)Cc;
    size_t stride = (size_t)gridDim.x * blockDim.x;
    for (size_t t = blockIdx.x * (size_t)blockDim.x + threadIdx.x; t < total; t += stride) {
        int c = (int)(t % Cc);
        float v = in[t] + b[c];
        v = v > 0.0f ? v : (__expf(v) - 1.0f);   // elu
        out[t] = (_Float16)v;
    }
}

// ---------------------------------------------------------------- final lin2 dot
__global__ void k_final(const float* __restrict__ h3, const float* __restrict__ w,
                        const float* __restrict__ b, float* __restrict__ out, int n, int dim) {
    int t = blockIdx.x * blockDim.x + threadIdx.x;
    if (t >= n) return;
    const float* hp = h3 + (size_t)t * dim;
    float s = 0.f;
    for (int i = 0; i < dim; ++i) s += hp[i] * w[i];
    out[t] = s + b[0];
}

// ================================================================ launcher
extern "C" void kernel_launch(void* const* d_in, const int* in_sizes, int n_in,
                              void* d_out, int out_size, void* d_ws, size_t ws_size,
                              hipStream_t stream) {
    (void)in_sizes; (void)n_in; (void)out_size; (void)ws_size;

    const float*     x        = (const float*)d_in[0];
    const long long* ei       = (const long long*)d_in[1];   // int64
    const float*     W1       = (const float*)d_in[2];
    const float*     att_src1 = (const float*)d_in[3];
    const float*     att_dst1 = (const float*)d_in[4];
    const float*     b1       = (const float*)d_in[5];
    const float*     W2       = (const float*)d_in[6];
    const float*     att_src2 = (const float*)d_in[7];
    const float*     att_dst2 = (const float*)d_in[8];
    const float*     b2       = (const float*)d_in[9];
    const float*     lin1_w   = (const float*)d_in[10];
    const float*     lin1_b   = (const float*)d_in[11];
    const float*     lin2_w   = (const float*)d_in[12];
    const float*     lin2_b   = (const float*)d_in[13];
    float* out = (float*)d_out;

    // ---- carve workspace (256B aligned) ----
    char* wsp = (char*)d_ws;
    auto alloc = [&](size_t bytes) -> void* {
        void* p = (void*)wsp;
        wsp += (bytes + 255) & ~((size_t)255);
        return p;
    };
    int*      src   = (int*)alloc((size_t)E_TOT * 4);
    int*      dst   = (int*)alloc((size_t)E_TOT * 4);
    _Float16* x16   = (_Float16*)alloc((size_t)N_NODES * IN_DIM * 2);   // reused as h3 (f32, same bytes)
    _Float16* Bp1   = (_Float16*)alloc((size_t)IN_DIM * C1 * 2);        // packed W1
    _Float16* Bp2   = (_Float16*)alloc((size_t)C1 * HID * 2);           // packed W2
    _Float16* Bp3   = (_Float16*)alloc((size_t)HID * HID * 2);          // packed lin1_w
    float*    h1    = (float*)alloc((size_t)N_NODES * C1 * 4);          // reused as h2
    float*    es1   = (float*)alloc((size_t)N_NODES * HEADS * 4);
    float*    ed1   = (float*)alloc((size_t)N_NODES * HEADS * 4);
    float*    sm1   = (float*)alloc((size_t)N_NODES * HEADS * 4);
    float*    dn1   = (float*)alloc((size_t)N_NODES * HEADS * 4);
    float*    elog  = (float*)alloc((size_t)E_TOT * HEADS * 4);
    float*    agg1  = (float*)alloc((size_t)N_NODES * C1 * 4);          // reused as agg2
    _Float16* h1a16 = (_Float16*)alloc((size_t)N_NODES * C1 * 2);       // reused as h2a16

    float*    h2    = h1;
    float*    agg2  = agg1;
    _Float16* h2a16 = h1a16;
    float*    h3    = (float*)x16;   // 100000*128*2B == 100000*64*4B

    const int TB = 256;
    auto cdiv = [](long long a, long long b) -> int { return (int)((a + b - 1) / b); };
    const int tilesM = N_NODES / 16;                  // 6250 = 5 * 1250
    const int gridM  = cdiv(tilesM / MTILES, 8);      // waves of MTILES tiles, 8 waves/block

    // ---- edges + converts + weight packing ----
    k_build_edges<<<cdiv(E_TOT, TB), TB, 0, stream>>>(ei, src, dst);
    k_f32_to_f16<<<2048, TB, 0, stream>>>(x, x16, (size_t)N_NODES * IN_DIM);
    k_pack_b<<<cdiv(IN_DIM * C1, TB), TB, 0, stream>>>(W1,     Bp1, IN_DIM, C1);
    k_pack_b<<<cdiv(C1 * HID,   TB), TB, 0, stream>>>(W2,     Bp2, C1,    HID);
    k_pack_b<<<cdiv(HID * HID,  TB), TB, 0, stream>>>(lin1_w, Bp3, HID,   HID);

    // ---- layer 1 ----
    {   // h1 = x @ W1 (bias applied post-aggregation, per reference)
        dim3 g(C1 / 16, gridM);
        k_wmma_gemm<IN_DIM><<<g, TB, 0, stream>>>(x16, Bp1, nullptr, h1, N_NODES, C1, 0, 0);
    }
    k_attn_logits<<<cdiv((long long)N_NODES * HEADS, TB), TB, 0, stream>>>(
        h1, att_src1, att_dst1, es1, ed1, N_NODES, HEADS, HID);
    k_fill_f32<<<1024, TB, 0, stream>>>(sm1, NEG_BIG, (size_t)N_NODES * HEADS);
    k_fill_f32<<<1024, TB, 0, stream>>>(dn1, 0.0f,    (size_t)N_NODES * HEADS);
    k_fill_f32<<<4096, TB, 0, stream>>>(agg1, 0.0f,   (size_t)N_NODES * C1);
    k_edge_logit_max<<<4096, TB, 0, stream>>>(src, dst, es1, ed1, elog, sm1, HEADS);
    k_edge_alpha_denom<<<4096, TB, 0, stream>>>(dst, elog, sm1, dn1, HEADS);
    k_edge_aggregate<<<8192, TB, 0, stream>>>(src, dst, elog, dn1, h1, agg1, HEADS, HID);
    k_bias_elu_f16<<<4096, TB, 0, stream>>>(agg1, b1, h1a16, (size_t)N_NODES, C1);

    // ---- layer 2 (1 head) ----
    {   // h2 = elu(out1) @ W2
        dim3 g(HID / 16, gridM);
        k_wmma_gemm<C1><<<g, TB, 0, stream>>>(h1a16, Bp2, nullptr, h2, N_NODES, HID, 0, 0);
    }
    k_attn_logits<<<cdiv(N_NODES, TB), TB, 0, stream>>>(
        h2, att_src2, att_dst2, es1, ed1, N_NODES, 1, HID);
    k_fill_f32<<<512,  TB, 0, stream>>>(sm1, NEG_BIG, (size_t)N_NODES);
    k_fill_f32<<<512,  TB, 0, stream>>>(dn1, 0.0f,    (size_t)N_NODES);
    k_fill_f32<<<2048, TB, 0, stream>>>(agg2, 0.0f,   (size_t)N_NODES * HID);
    k_edge_logit_max<<<4096, TB, 0, stream>>>(src, dst, es1, ed1, elog, sm1, 1);
    k_edge_alpha_denom<<<4096, TB, 0, stream>>>(dst, elog, sm1, dn1, 1);
    k_edge_aggregate<<<8192, TB, 0, stream>>>(src, dst, elog, dn1, h2, agg2, 1, HID);
    k_bias_elu_f16<<<2048, TB, 0, stream>>>(agg2, b2, h2a16, (size_t)N_NODES, HID);

    // ---- MLP head ----
    {   // h3 = relu(elu(out2) @ lin1_w + lin1_b), bias+relu fused
        dim3 g(HID / 16, gridM);
        k_wmma_gemm<HID><<<g, TB, 0, stream>>>(h2a16, Bp3, lin1_b, h3, N_NODES, HID, 1, 1);
    }
    k_final<<<cdiv(N_NODES, TB), TB, 0, stream>>>(h3, lin2_w, lin2_b, out, N_NODES, HID);
}
